// AttentionCostVolume_78451872629269
// MI455X (gfx1250) — compile-verified, matrix operand order checked by reading, not run on hardware
//
#include <hip/hip_runtime.h>
#include <hip/hip_bf16.h>

// ---------------------------------------------------------------------------
// Problem constants (match reference)
// ---------------------------------------------------------------------------
#define BATCH 4
#define CH    96
#define IMH   160
#define IMW   320
#define MAXD  49

typedef __bf16 bf16x16 __attribute__((ext_vector_type(16)));
typedef float  f32x8   __attribute__((ext_vector_type(8)));
typedef int    v4i_vs  __attribute__((vector_size(16)));   // matches builtin sig

union Frag { bf16x16 v; unsigned int u[8]; };
union Acc  { f32x8 v; float f[8]; };

// Hardware RNE convert (lowers to v_cvt *bf16* on gfx1250)
__device__ __forceinline__ unsigned short f2bf(float f) {
  union { __bf16 b; unsigned short s; } cv;
  cv.b = (__bf16)f;
  return cv.s;
}

#define WMMA_BF16(A, B, C) \
  __builtin_amdgcn_wmma_f32_16x16x32_bf16(false, (A), false, (B), (short)0, (C), false, false)

#define GLOBAL_AS __attribute__((address_space(1)))
#define LDS_AS    __attribute__((address_space(3)))

#if defined(__gfx1250__) && __has_builtin(__builtin_amdgcn_global_load_async_to_lds_b128)
#define HAS_ASYNC_LDS 1
#else
#define HAS_ASYNC_LDS 0
#endif

__device__ __forceinline__ void wait_asynccnt0() {
#if __has_builtin(__builtin_amdgcn_s_wait_asynccnt)
  __builtin_amdgcn_s_wait_asynccnt(0);
#else
  asm volatile("s_wait_asynccnt 0x0" ::: "memory");
#endif
}

// ---------------------------------------------------------------------------
// Prep: fold BN into scale/shift, repack weights into bf16 WMMA-friendly order
//   WB[tap][o][c]  (tap = kh*3+kw),  WD[o][c]
// ---------------------------------------------------------------------------
__global__ void prep_kernel(const float* __restrict__ conv_w,
                            const float* __restrict__ gamma,
                            const float* __restrict__ beta,
                            const float* __restrict__ mean,
                            const float* __restrict__ var,
                            const float* __restrict__ desc_w,
                            unsigned short* __restrict__ WB,
                            unsigned short* __restrict__ WD,
                            float* __restrict__ scaleArr,
                            float* __restrict__ shiftArr) {
  int i = blockIdx.x * 256 + threadIdx.x;
  if (i < 9 * CH * CH) {
    int tap = i / (CH * CH);
    int rem = i - tap * (CH * CH);
    int o = rem / CH, c = rem - o * CH;
    WB[i] = f2bf(conv_w[(o * CH + c) * 9 + tap]);   // (o,c,kh,kw) -> (tap,o,c)
  }
  if (i < CH * CH) WD[i] = f2bf(desc_w[i]);         // already [o][c]
  if (i < CH) {
    float s = gamma[i] * rsqrtf(var[i] + 1e-5f);
    scaleArr[i] = s;
    shiftArr[i] = beta[i] - mean[i] * s;
  }
}

// ---------------------------------------------------------------------------
// Feature kernel: conv3x3 (implicit GEMM, bf16 WMMA) + BN + LeakyReLU
//                 + 1x1 desc GEMM (bf16 WMMA) + bias + L2 norm.
// Block = 128 threads (4 waves). Tile = 64 pixels (one h row) x 96 channels.
// Grid  = (W/64, H, 2*BATCH)   z encodes image (z>>2) and batch (z&3).
// Descriptors stored bf16, layout [b][h][w][c] (c contiguous).
// ---------------------------------------------------------------------------
__global__ __launch_bounds__(128)
void feat_kernel(const float* __restrict__ imL, const float* __restrict__ imR,
                 const unsigned short* __restrict__ WB,
                 const unsigned short* __restrict__ WD,
                 const float* __restrict__ scaleArr,
                 const float* __restrict__ shiftArr,
                 const float* __restrict__ desc_b,
                 unsigned short* __restrict__ descX,
                 unsigned short* __restrict__ descY) {
  const int tid = threadIdx.x;
  const int w0g = blockIdx.x * 64;
  const int h   = blockIdx.y;
  const int z   = blockIdx.z;
  const int img = z >> 2;
  const int b   = z & 3;
  const float* im = img ? imR : imL;
  unsigned short* dsc = img ? descY : descX;

  __shared__ __align__(16) unsigned short sIn[3 * 66 * CH]; // [row][wl][c]
  __shared__ __align__(16) unsigned short sZ[64 * CH];      // conv out tile

  // ---- stage 3 input rows (h-1..h+1), w range [w0g-1, w0g+64], to LDS bf16
  for (int idx = tid; idx < 3 * 66 * CH; idx += 128) {
    int wl  = idx % 66;
    int rc  = idx / 66;
    int c   = rc % CH;
    int row = rc / CH;
    int hh = h + row - 1;
    int ww = w0g + wl - 1;
    float v = 0.0f;
    if (hh >= 0 && hh < IMH && ww >= 0 && ww < IMW)
      v = im[((b * CH + c) * IMH + hh) * IMW + ww];
    sIn[(row * 66 + wl) * CH + c] = f2bf(v);
  }
  __syncthreads();

  const int wv   = tid >> 5;        // wave id: m-subtile (16 pixels)
  const int L    = tid & 31;
  const int ln   = L & 15;
  const int kOff = (L >> 4) << 3;   // 0 or 8: K/M offset for upper lanes
  const int pw   = wv * 16;

  int kk[8];                         // per-lane K offsets for fragment pairs
#pragma unroll
  for (int p = 0; p < 8; ++p)
    kk[p] = ((p & 4) << 2) + kOff + ((p & 3) << 1);

  // ---- conv3x3 as 9 shifted GEMMs, K = 96 channels per tap (3 chunks of 32)
  f32x8 acc[6];
#pragma unroll
  for (int nt = 0; nt < 6; ++nt) acc[nt] = {};

  for (int tap = 0; tap < 9; ++tap) {
    const int kh = tap / 3, kw = tap % 3;
    const unsigned short* arow = &sIn[((kh * 66) + (pw + ln + kw)) * CH];
    const unsigned short* wtap = WB + tap * (CH * CH) + ln * CH;
#pragma unroll
    for (int cc = 0; cc < 3; ++cc) {
      const int c0 = cc * 32;
      Frag a;
#pragma unroll
      for (int p = 0; p < 8; ++p)
        a.u[p] = *(const unsigned int*)(arow + c0 + kk[p]);
#pragma unroll
      for (int nt = 0; nt < 6; ++nt) {
        const unsigned short* wp = wtap + nt * 16 * CH + c0;
        Frag bb;
#pragma unroll
        for (int p = 0; p < 8; ++p)
          bb.u[p] = *(const unsigned int*)(wp + kk[p]);
        acc[nt] = WMMA_BF16(a.v, bb.v, acc[nt]);
      }
    }
  }

  // ---- BN (folded) + LeakyReLU, spill z tile to LDS as bf16 (A-layout input)
#pragma unroll
  for (int nt = 0; nt < 6; ++nt) {
    const int ch = nt * 16 + ln;
    const float s  = scaleArr[ch];
    const float sh = shiftArr[ch];
    Acc u; u.v = acc[nt];
#pragma unroll
    for (int r = 0; r < 8; ++r) {
      float v = u.f[r] * s + sh;
      v = (v >= 0.0f) ? v : 0.01f * v;
      sZ[(pw + kOff + r) * CH + ch] = f2bf(v);
    }
  }
  __syncthreads();

  // ---- 1x1 desc GEMM: d[16,96] = z[16,96] x WD^T
  f32x8 dacc[6];
#pragma unroll
  for (int nt = 0; nt < 6; ++nt) dacc[nt] = {};

  const unsigned short* zrow = &sZ[(pw + ln) * CH];
#pragma unroll
  for (int cc = 0; cc < 3; ++cc) {
    const int c0 = cc * 32;
    Frag a;
#pragma unroll
    for (int p = 0; p < 8; ++p)
      a.u[p] = *(const unsigned int*)(zrow + c0 + kk[p]);
#pragma unroll
    for (int nt = 0; nt < 6; ++nt) {
      const unsigned short* wp = WD + (nt * 16 + ln) * CH + c0;
      Frag bb;
#pragma unroll
      for (int p = 0; p < 8; ++p)
        bb.u[p] = *(const unsigned int*)(wp + kk[p]);
      dacc[nt] = WMMA_BF16(a.v, bb.v, dacc[nt]);
    }
  }

  // ---- bias + per-pixel L2 norm (reduce across 16 lanes holding channels)
  float df[6][8];
  float ss[8];
#pragma unroll
  for (int r = 0; r < 8; ++r) ss[r] = 0.0f;
#pragma unroll
  for (int nt = 0; nt < 6; ++nt) {
    const float bias = desc_b[nt * 16 + ln];
    Acc u; u.v = dacc[nt];
#pragma unroll
    for (int r = 0; r < 8; ++r) {
      float v = u.f[r] + bias;
      df[nt][r] = v;
      ss[r] += v * v;
    }
  }
#pragma unroll
  for (int r = 0; r < 8; ++r) {
    float t = ss[r];
    t += __shfl_xor(t, 1, 16);
    t += __shfl_xor(t, 2, 16);
    t += __shfl_xor(t, 4, 16);
    t += __shfl_xor(t, 8, 16);
    ss[r] = rsqrtf(t);
  }
  const long rowBase = ((long)(b * IMH + h) * IMW + (w0g + pw + kOff)) * CH;
#pragma unroll
  for (int nt = 0; nt < 6; ++nt) {
    const int ch = nt * 16 + ln;
#pragma unroll
    for (int r = 0; r < 8; ++r)
      dsc[rowBase + (long)r * CH + ch] = f2bf(df[nt][r] * ss[r]);
  }
}

// ---------------------------------------------------------------------------
// Cost volume: per wave, X block [16 x 96] against Y window [96 x 64]
// (columns w' in [w0-48, w0+15]) -> 16x64 correlation, keep band d=w-w' in
// [0,48]. Y window staged in LDS via CDNA5 async-to-LDS copies (zero-filled
// for w'<0 == reference left padding). Block = 128 (4 waves), 64 px of a row.
// ---------------------------------------------------------------------------
__global__ __launch_bounds__(128)
void cost_kernel(const unsigned short* __restrict__ descX,
                 const unsigned short* __restrict__ descY,
                 float* __restrict__ out) {
  const int tid = threadIdx.x;
  const int w0g = blockIdx.x * 64;
  const int h   = blockIdx.y;
  const int b   = blockIdx.z;

  __shared__ __align__(16) unsigned short sY[112 * CH]; // w' in [w0g-48, w0g+63]

  const long yRow0 = (long)(b * IMH + h) * IMW;  // element row base into descY

#if HAS_ASYNC_LDS
  // 112 rows x 192B; each row = 12 x 16B chunks. Async-copy in-range rows,
  // zero-store out-of-range (w' < 0) rows directly.
  for (int idx = tid; idx < 112 * 12; idx += 128) {
    int wl = idx / 12;
    int ck = idx - wl * 12;
    int wq = w0g - 48 + wl;
    unsigned short* ldst = &sY[wl * CH + ck * 8];
    if (wq >= 0) {
      const unsigned short* gsrc = descY + (yRow0 + wq) * CH + ck * 8;
      __builtin_amdgcn_global_load_async_to_lds_b128(
          (GLOBAL_AS v4i_vs*)gsrc, (LDS_AS v4i_vs*)ldst, 0, 0);
    } else {
      *(uint4*)ldst = make_uint4(0u, 0u, 0u, 0u);
    }
  }
  wait_asynccnt0();
  __syncthreads();
#else
  for (int idx = tid; idx < 112 * CH; idx += 128) {
    int c  = idx % CH;
    int wl = idx / CH;
    int wq = w0g - 48 + wl;
    unsigned short v = 0;
    if (wq >= 0) v = descY[(yRow0 + wq) * CH + c];
    sY[wl * CH + c] = v;
  }
  __syncthreads();
#endif

  const int wv   = tid >> 5;
  const int L    = tid & 31;
  const int ln   = L & 15;
  const int kOff = (L >> 4) << 3;
  const int w0   = w0g + wv * 16;

  int kk[8];
#pragma unroll
  for (int p = 0; p < 8; ++p)
    kk[p] = ((p & 4) << 2) + kOff + ((p & 3) << 1);

  const unsigned short* xrow = descX + (yRow0 + (w0 + ln)) * CH;

  f32x8 cacc[4];
#pragma unroll
  for (int nt = 0; nt < 4; ++nt) cacc[nt] = {};

#pragma unroll
  for (int cc = 0; cc < 3; ++cc) {
    const int c0 = cc * 32;
    Frag a;
#pragma unroll
    for (int p = 0; p < 8; ++p)
      a.u[p] = *(const unsigned int*)(xrow + c0 + kk[p]);
#pragma unroll
    for (int nt = 0; nt < 4; ++nt) {
      const unsigned short* brow = &sY[(wv * 16 + nt * 16 + ln) * CH + c0];
      Frag bb;
#pragma unroll
      for (int p = 0; p < 8; ++p)
        bb.u[p] = *(const unsigned int*)(brow + kk[p]);
      cacc[nt] = WMMA_BF16(a.v, bb.v, cacc[nt]);
    }
  }

  // keep diagonal band: d = w - w' = 48 + kOff + r - 16*nt - ln
#pragma unroll
  for (int nt = 0; nt < 4; ++nt) {
    Acc u; u.v = cacc[nt];
#pragma unroll
    for (int r = 0; r < 8; ++r) {
      int w  = w0 + kOff + r;
      int dd = 48 + kOff + r - nt * 16 - ln;
      if (dd >= 0 && dd < MAXD)
        out[((long)(dd * BATCH + b) * IMH + h) * IMW + w] = u.f[r];
    }
  }
}

// ---------------------------------------------------------------------------
// Launcher
// ---------------------------------------------------------------------------
extern "C" void kernel_launch(void* const* d_in, const int* in_sizes, int n_in,
                              void* d_out, int out_size, void* d_ws, size_t ws_size,
                              hipStream_t stream) {
  const float* imL    = (const float*)d_in[0];
  const float* imR    = (const float*)d_in[1];
  const float* conv_w = (const float*)d_in[2];
  const float* gamma  = (const float*)d_in[3];
  const float* beta   = (const float*)d_in[4];
  const float* mean   = (const float*)d_in[5];
  const float* var    = (const float*)d_in[6];
  const float* desc_w = (const float*)d_in[7];
  const float* desc_b = (const float*)d_in[8];
  float* out = (float*)d_out;

  // Workspace carve-out (all offsets 16B-aligned; base is page-aligned)
  char* ws = (char*)d_ws;
  unsigned short* WB = (unsigned short*)(ws);                       // 9*96*96*2 = 165,888
  unsigned short* WD = (unsigned short*)(ws + 165888);              // 96*96*2   = 18,432
  float* scaleArr    = (float*)(ws + 165888 + 18432);               // 384
  float* shiftArr    = (float*)(ws + 165888 + 18432 + 384);         // 384
  unsigned short* descX = (unsigned short*)(ws + 185088);           // 4*160*320*96*2 = 39,321,600
  unsigned short* descY = (unsigned short*)(ws + 185088 + 39321600);

  prep_kernel<<<dim3((9 * CH * CH + 255) / 256), dim3(256), 0, stream>>>(
      conv_w, gamma, beta, mean, var, desc_w, WB, WD, scaleArr, shiftArr);

  feat_kernel<<<dim3(IMW / 64, IMH, 2 * BATCH), dim3(128), 0, stream>>>(
      imL, imR, WB, WD, scaleArr, shiftArr, desc_b, descX, descY);

  cost_kernel<<<dim3(IMW / 64, IMH, BATCH), dim3(128), 0, stream>>>(
      descX, descY, out);

  (void)in_sizes; (void)n_in; (void)out_size; (void)ws_size;
}
